// TPAR_79053168050535
// MI455X (gfx1250) — compile-verified, matrix operand order checked by reading, not run on hardware
//
#include <hip/hip_runtime.h>
#include <hip/hip_bf16.h>
#include <math.h>

// ---------------------------------------------------------------------------
// Problem constants (match reference)
// ---------------------------------------------------------------------------
#define LNUM   3
#define DDIM   128
#define ADIM   64
#define NREL2  401
#define NNODE  100000
#define NEDGE  600000
#define NQC    512
#define NENT   50000

// Packed-weight tile bookkeeping: each WMMA B tile (32x16 bf16) is stored as
// 32 lanes x 8 dwords = 256 u32, laid out so a lane fetches its fragment with
// two 16B loads at (tile_base + lane*8).
#define TILE_U32        256
#define OFF_WH          64              // attn: 64 tiles (K=512 x N=64)
#define OFF_WIH         96              // Wh:   32 tiles (K=128 x N=128)
#define OFF_WHH         192             // Wih:  96 tiles (K=128 x N=384)
#define TILES_PER_LAYER 288             // Whh:  96 tiles

typedef __attribute__((ext_vector_type(16))) __bf16 v16bf;
typedef __attribute__((ext_vector_type(8)))  float  v8f;

struct FragBF { union { unsigned int u[8]; v16bf v; }; };
struct AccF   { union { float f[8];        v8f   v; }; };

// Native f32->bf16 convert pair; compiler lowers to packed bf16 converts.
__device__ __forceinline__ unsigned int pk2bf(float a, float b) {
  union { __bf16 h[2]; unsigned int u; } r;
  r.h[0] = (__bf16)a;
  r.h[1] = (__bf16)b;
  return r.u;
}

__device__ __forceinline__ float sigmoidf_(float x) { return 1.0f / (1.0f + expf(-x)); }

// A fragment (16x32 bf16): lane holds row (lane&15); K halves selected by
// hi=lane>>4 per ISA layout: u[0..3] = K {hi*8 .. hi*8+7},
// u[4..7] = K {16+hi*8 .. 16+hi*8+7}.  p = row base + ktile*32.
__device__ __forceinline__ FragBF load_row_frag(const float* p, int hi) {
  FragBF a;
  const float4* p0 = (const float4*)(p + hi * 8);
  const float4* p1 = (const float4*)(p + 16 + hi * 8);
  float4 x0 = p0[0], x1 = p0[1], y0 = p1[0], y1 = p1[1];
  a.u[0] = pk2bf(x0.x, x0.y); a.u[1] = pk2bf(x0.z, x0.w);
  a.u[2] = pk2bf(x1.x, x1.y); a.u[3] = pk2bf(x1.z, x1.w);
  a.u[4] = pk2bf(y0.x, y0.y); a.u[5] = pk2bf(y0.z, y0.w);
  a.u[6] = pk2bf(y1.x, y1.y); a.u[7] = pk2bf(y1.z, y1.w);
  return a;
}

__device__ __forceinline__ FragBF load_b_frag(const unsigned int* tile_base, int lane) {
  FragBF b;
  const uint4* bp = (const uint4*)(tile_base + lane * 8);
  uint4 b0 = bp[0], b1 = bp[1];
  b.u[0] = b0.x; b.u[1] = b0.y; b.u[2] = b0.z; b.u[3] = b0.w;
  b.u[4] = b1.x; b.u[5] = b1.y; b.u[6] = b1.z; b.u[7] = b1.w;
  return b;
}

// h_hau fragment built on the fly: h(d) = wt1*delta + bt1 + sin(wt2*delta + bt2)
__device__ __forceinline__ FragBF hhau_frag(const float* wt1l, const float* bt1l,
                                            const float* wt2l, const float* bt2l,
                                            int d0, float delta) {
  FragBF a;
  float v0[8], v1[8];
#pragma unroll
  for (int t = 0; t < 8; ++t) {
    int d = d0 + t;
    v0[t] = wt1l[d] * delta + bt1l[d] + sinf(wt2l[d] * delta + bt2l[d]);
    d = d0 + 16 + t;
    v1[t] = wt1l[d] * delta + bt1l[d] + sinf(wt2l[d] * delta + bt2l[d]);
  }
  a.u[0] = pk2bf(v0[0], v0[1]); a.u[1] = pk2bf(v0[2], v0[3]);
  a.u[2] = pk2bf(v0[4], v0[5]); a.u[3] = pk2bf(v0[6], v0[7]);
  a.u[4] = pk2bf(v1[0], v1[1]); a.u[5] = pk2bf(v1[2], v1[3]);
  a.u[6] = pk2bf(v1[4], v1[5]); a.u[7] = pk2bf(v1[6], v1[7]);
  return a;
}

__device__ __forceinline__ v8f wmma_bf16(const FragBF& a, const FragBF& b, v8f c) {
  return __builtin_amdgcn_wmma_f32_16x16x32_bf16(false, a.v, false, b.v, (short)0, c,
                                                 false, false);
}

// ---------------------------------------------------------------------------
// Utility kernels
// ---------------------------------------------------------------------------
__global__ void zero_kernel(float* __restrict__ p, size_t n) {
  size_t i = (size_t)blockIdx.x * blockDim.x + threadIdx.x;
  size_t st = (size_t)gridDim.x * blockDim.x;
  for (; i < n; i += st) p[i] = 0.0f;
}

// Repack every weight matrix into per-lane WMMA B-fragment layout (bf16 pairs).
// B[k][n] = W[n][k] because the reference computes X @ W^T.
__global__ void prep_weights(const float* __restrict__ Ws, const float* __restrict__ Wr,
                             const float* __restrict__ Wqr_w, const float* __restrict__ Wtau,
                             const float* __restrict__ Wh, const float* __restrict__ Wih,
                             const float* __restrict__ Whh, unsigned int* __restrict__ wpack) {
  int idx = blockIdx.x * blockDim.x + threadIdx.x;
  const int total = LNUM * TILES_PER_LAYER * TILE_U32;
  if (idx >= total) return;
  int layer = idx / (TILES_PER_LAYER * TILE_U32);
  int rem   = idx % (TILES_PER_LAYER * TILE_U32);
  int tile  = rem / TILE_U32;
  int wi    = rem % TILE_U32;
  int lane = wi >> 3, j = wi & 7;
  int hi = lane >> 4, n16 = lane & 15;
  int kb = ((j < 4) ? (2 * j) : (16 + 2 * (j - 4))) + hi * 8;   // even, pair kb,kb+1
  float b0, b1;
  if (tile < OFF_WH) {                       // attn concat: [Ws | Wr | Wqr | Wtau]
    int kt = tile >> 2, nt = tile & 3;
    int k = kt * 32 + kb, n = nt * 16 + n16;
    int seg = k >> 7, kk = k & 127;
    const float* W = (seg == 0) ? Ws : (seg == 1) ? Wr : (seg == 2) ? Wqr_w : Wtau;
    const float* p = W + ((size_t)layer * ADIM + n) * DDIM + kk;
    b0 = p[0]; b1 = p[1];
  } else if (tile < OFF_WIH) {               // Wh
    int t = tile - OFF_WH, kt = t >> 3, nt = t & 7;
    const float* p = Wh + ((size_t)layer * DDIM + nt * 16 + n16) * DDIM + kt * 32 + kb;
    b0 = p[0]; b1 = p[1];
  } else {                                   // Wih / Whh (3D x D)
    const float* W = (tile < OFF_WHH) ? Wih : Whh;
    int t = tile - ((tile < OFF_WHH) ? OFF_WIH : OFF_WHH);
    int kt = t / 24, nt = t % 24;
    const float* p = W + ((size_t)layer * (3 * DDIM) + nt * 16 + n16) * DDIM + kt * 32 + kb;
    b0 = p[0]; b1 = p[1];
  }
  wpack[idx] = pk2bf(b0, b1);
}

// ---------------------------------------------------------------------------
// Edge kernel: 2 waves/block, one 32-edge tile per wave (two 16-row A tiles
// sharing every B fragment -> 8 WMMAs per B-fetch group). K loop fully
// unrolled: segment dispatch folds to straight-line code, no loop-carried
// fragment copies.
// ---------------------------------------------------------------------------
__global__ void __launch_bounds__(64)
edge_kernel(int layer, const float* __restrict__ rela, const float* __restrict__ wqr_b,
            const float* __restrict__ walpha_w, const float* __restrict__ walpha_b,
            const float* __restrict__ wt1, const float* __restrict__ bt1,
            const float* __restrict__ wt2, const float* __restrict__ bt2,
            const int* __restrict__ q_rel, const int* __restrict__ q_tau,
            const int* __restrict__ e_ridx, const int* __restrict__ e_rel,
            const int* __restrict__ e_tau, const int* __restrict__ e_sub,
            const int* __restrict__ e_obj, const float* __restrict__ state,
            float* __restrict__ agg, const unsigned int* __restrict__ wpack) {
  __shared__ float sh_alpha[2][32];
  __shared__ float sh_delta[2][32];
  __shared__ int   sh_sub[2][32], sh_rel[2][32], sh_qr[2][32], sh_obj[2][32];

  const int lane = threadIdx.x & 31;
  const int w    = threadIdx.x >> 5;
  const int tile = blockIdx.x * 2 + w;          // 18750 tiles exactly
  const int row = lane & 15, hi = lane >> 4;

  const size_t ebase = (size_t)layer * NEDGE + (size_t)tile * 32;
  const size_t e = ebase + lane;                // one edge per lane
  // prefetch next tile's index stream (global_prefetch_b8)
  __builtin_prefetch(e_sub + e + 64, 0, 1);
  __builtin_prefetch(e_obj + e + 64, 0, 1);

  const int sub = e_sub[e], rel = e_rel[e], ridx = e_ridx[e], tau = e_tau[e];
  const int tq = q_tau[ridx];
  sh_sub[w][lane]   = sub;
  sh_rel[w][lane]   = rel;
  sh_obj[w][lane]   = e_obj[e];
  sh_qr[w][lane]    = q_rel[ridx];
  sh_delta[w][lane] = (tau >= 0) ? (float)(tau - tq) : 0.0f;
  __syncthreads();

  // per-lane row pointers for the two A tiles (edges row and row+16)
  const float* relL = rela + (size_t)layer * NREL2 * DDIM;
  const float* seg0[2] = { state + (size_t)sh_sub[w][row] * DDIM,
                           state + (size_t)sh_sub[w][row + 16] * DDIM };
  const float* seg1[2] = { relL + (size_t)sh_rel[w][row] * DDIM,
                           relL + (size_t)sh_rel[w][row + 16] * DDIM };
  const float* seg2[2] = { relL + (size_t)sh_qr[w][row] * DDIM,
                           relL + (size_t)sh_qr[w][row + 16] * DDIM };
  const float dl0 = sh_delta[w][row], dl1 = sh_delta[w][row + 16];

  const float* wt1l = wt1 + (size_t)layer * DDIM;
  const float* bt1l = bt1 + (size_t)layer * DDIM;
  const float* wt2l = wt2 + (size_t)layer * DDIM;
  const float* bt2l = bt2 + (size_t)layer * DDIM;
  const unsigned int* wl = wpack + (size_t)layer * TILES_PER_LAYER * TILE_U32;

  AccF accA[4], accB[4];
#pragma unroll
  for (int nt = 0; nt < 4; ++nt)
#pragma unroll
    for (int j = 0; j < 8; ++j) { accA[nt].f[j] = 0.0f; accB[nt].f[j] = 0.0f; }

#pragma unroll
  for (int kt = 0; kt < 16; ++kt) {
    FragBF a0, a1;
    const int seg = kt >> 2, ktl = kt & 3;     // compile-time after unroll
    if (seg == 0) {
      a0 = load_row_frag(seg0[0] + ktl * 32, hi);
      a1 = load_row_frag(seg0[1] + ktl * 32, hi);
    } else if (seg == 1) {
      a0 = load_row_frag(seg1[0] + ktl * 32, hi);
      a1 = load_row_frag(seg1[1] + ktl * 32, hi);
    } else if (seg == 2) {
      a0 = load_row_frag(seg2[0] + ktl * 32, hi);
      a1 = load_row_frag(seg2[1] + ktl * 32, hi);
    } else {
      const int d0 = ktl * 32 + hi * 8;
      a0 = hhau_frag(wt1l, bt1l, wt2l, bt2l, d0, dl0);
      a1 = hhau_frag(wt1l, bt1l, wt2l, bt2l, d0, dl1);
    }
#pragma unroll
    for (int nt = 0; nt < 4; ++nt) {
      FragBF b = load_b_frag(wl + (size_t)(kt * 4 + nt) * TILE_U32, lane);
      accA[nt].v = wmma_bf16(a0, b, accA[nt].v);
      accB[nt].v = wmma_bf16(a1, b, accB[nt].v);
    }
  }

  // bias + relu + attn . walpha, cross-lane reduce within each 16-lane half
  float sA[8], sB[8];
#pragma unroll
  for (int j = 0; j < 8; ++j) {
    float a = 0.0f, b = 0.0f;
#pragma unroll
    for (int nt = 0; nt < 4; ++nt) {
      const int col = nt * 16 + row;
      const float bias = wqr_b[(size_t)layer * ADIM + col];
      const float ww   = walpha_w[(size_t)layer * ADIM + col];
      a += fmaxf(accA[nt].f[j] + bias, 0.0f) * ww;
      b += fmaxf(accB[nt].f[j] + bias, 0.0f) * ww;
    }
    sA[j] = a; sB[j] = b;
  }
#pragma unroll
  for (int m = 1; m < 16; m <<= 1)
#pragma unroll
    for (int j = 0; j < 8; ++j) {
      sA[j] += __shfl_xor(sA[j], m, 32);
      sB[j] += __shfl_xor(sB[j], m, 32);
    }
  if (row == 0) {
    const float wb = walpha_b[layer];
#pragma unroll
    for (int j = 0; j < 8; ++j) {
      sh_alpha[w][hi * 8 + j]      = sigmoidf_(sA[j] + wb);
      sh_alpha[w][16 + hi * 8 + j] = sigmoidf_(sB[j] + wb);
    }
  }
  __syncthreads();

  // scatter: agg[obj] += alpha * (hs + hr + h_hau)
  for (int el = 0; el < 32; ++el) {
    const float al = sh_alpha[w][el];
    const float dl = sh_delta[w][el];
    const float* hsp = state + (size_t)sh_sub[w][el] * DDIM;
    const float* hrp = relL + (size_t)sh_rel[w][el] * DDIM;
    float* ap = agg + (size_t)sh_obj[w][el] * DDIM;
#pragma unroll
    for (int c = 0; c < 4; ++c) {
      const int d = c * 32 + lane;
      const float hh = wt1l[d] * dl + bt1l[d] + sinf(wt2l[d] * dl + bt2l[d]);
      unsafeAtomicAdd(ap + d, al * (hsp[d] + hrp[d] + hh));  // global_atomic_add_f32
    }
  }
}

// ---------------------------------------------------------------------------
// Node kernel: 2 waves/block, one 16-node tile per wave.
//   hidden = relu(agg @ Wh^T)      (32 WMMAs) -> staged in LDS
//   GRU(hidden, h0)                (192 WMMAs), state updated in place.
// ---------------------------------------------------------------------------
__global__ void __launch_bounds__(64)
node_kernel(int layer, const float* __restrict__ agg, float* __restrict__ state,
            const unsigned int* __restrict__ wpack, const float* __restrict__ bih,
            const float* __restrict__ bhh) {
  __shared__ float sh_h[2][16][DDIM];    // relu(agg@Wh^T) tile
  __shared__ float sh_h0[2][16][DDIM];   // staged h0 (avoids RAW on in-place state)

  const int lane = threadIdx.x & 31;
  const int w    = threadIdx.x >> 5;
  const int tile = blockIdx.x * 2 + w;   // 6250 tiles exactly
  const int row = lane & 15, hi = lane >> 4;
  const size_t nodeBase = (size_t)tile * 16;

  const unsigned int* wl   = wpack + (size_t)layer * TILES_PER_LAYER * TILE_U32;
  const unsigned int* whW  = wl + (size_t)OFF_WH  * TILE_U32;
  const unsigned int* wihW = wl + (size_t)OFF_WIH * TILE_U32;
  const unsigned int* whhW = wl + (size_t)OFF_WHH * TILE_U32;

  // stage h0 tile into LDS
  {
    const float4* src = (const float4*)(state + nodeBase * DDIM);
    float4* dst = (float4*)&sh_h0[w][0][0];
#pragma unroll
    for (int i = 0; i < 16; ++i) dst[i * 32 + lane] = src[i * 32 + lane];
  }

  // phase 1: hidden = relu(agg @ Wh^T)
  AccF acc[8];
#pragma unroll
  for (int nt = 0; nt < 8; ++nt)
#pragma unroll
    for (int j = 0; j < 8; ++j) acc[nt].f[j] = 0.0f;
  const float* arow = agg + (nodeBase + row) * DDIM;
#pragma unroll
  for (int kt = 0; kt < 4; ++kt) {
    FragBF a = load_row_frag(arow + kt * 32, hi);
#pragma unroll
    for (int nt = 0; nt < 8; ++nt) {
      FragBF b = load_b_frag(whW + (size_t)(kt * 8 + nt) * TILE_U32, lane);
      acc[nt].v = wmma_bf16(a, b, acc[nt].v);
    }
  }
#pragma unroll
  for (int nt = 0; nt < 8; ++nt)
#pragma unroll
    for (int j = 0; j < 8; ++j)
      sh_h[w][j + 8 * hi][nt * 16 + row] = fmaxf(acc[nt].f[j], 0.0f);
  __syncthreads();

  // phase 2: GRU, one 16-column slab at a time (6 accumulators live)
  const float* bihl = bih + (size_t)layer * 3 * DDIM;
  const float* bhhl = bhh + (size_t)layer * 3 * DDIM;
  for (int nt = 0; nt < 8; ++nt) {
    AccF ir, iz, inn, hr, hz, hn;
#pragma unroll
    for (int j = 0; j < 8; ++j) {
      ir.f[j] = iz.f[j] = inn.f[j] = hr.f[j] = hz.f[j] = hn.f[j] = 0.0f;
    }
#pragma unroll
    for (int kt = 0; kt < 4; ++kt) {
      FragBF ax = load_row_frag(&sh_h[w][row][0] + kt * 32, hi);   // ds_load path
      FragBF ah = load_row_frag(&sh_h0[w][row][0] + kt * 32, hi);
      FragBF br = load_b_frag(wihW + (size_t)(kt * 24 + 0 * 8 + nt) * TILE_U32, lane);
      FragBF bz = load_b_frag(wihW + (size_t)(kt * 24 + 1 * 8 + nt) * TILE_U32, lane);
      FragBF bn = load_b_frag(wihW + (size_t)(kt * 24 + 2 * 8 + nt) * TILE_U32, lane);
      ir.v  = wmma_bf16(ax, br, ir.v);
      iz.v  = wmma_bf16(ax, bz, iz.v);
      inn.v = wmma_bf16(ax, bn, inn.v);
      FragBF cr = load_b_frag(whhW + (size_t)(kt * 24 + 0 * 8 + nt) * TILE_U32, lane);
      FragBF cz = load_b_frag(whhW + (size_t)(kt * 24 + 1 * 8 + nt) * TILE_U32, lane);
      FragBF cn = load_b_frag(whhW + (size_t)(kt * 24 + 2 * 8 + nt) * TILE_U32, lane);
      hr.v = wmma_bf16(ah, cr, hr.v);
      hz.v = wmma_bf16(ah, cz, hz.v);
      hn.v = wmma_bf16(ah, cn, hn.v);
    }
    const int col = nt * 16 + row;
    const float bir = bihl[col], biz = bihl[DDIM + col], bin = bihl[2 * DDIM + col];
    const float bhr = bhhl[col], bhz = bhhl[DDIM + col], bhn = bhhl[2 * DDIM + col];
#pragma unroll
    for (int j = 0; j < 8; ++j) {
      const int m = j + 8 * hi;
      const float r = sigmoidf_(ir.f[j] + bir + hr.f[j] + bhr);
      const float z = sigmoidf_(iz.f[j] + biz + hz.f[j] + bhz);
      const float nn = tanhf(inn.f[j] + bin + r * (hn.f[j] + bhn));
      const float h0v = sh_h0[w][m][col];
      state[(nodeBase + m) * DDIM + col] = (1.0f - z) * nn + z * h0v;
    }
  }
}

// ---------------------------------------------------------------------------
// Final kernel: scores = state @ Wfinal, scattered into [NQ, NENT]
// ---------------------------------------------------------------------------
__global__ void __launch_bounds__(128)
final_kernel(const float* __restrict__ state, const float* __restrict__ Wfinal,
             const int* __restrict__ nb, const int* __restrict__ ne,
             float* __restrict__ out) {
  const int lane = threadIdx.x & 31;
  const int w    = threadIdx.x >> 5;
  const size_t n = (size_t)blockIdx.x * 4 + w;
  if (n >= NNODE) return;
  const float4 a = ((const float4*)(state + n * DDIM))[lane];
  const float4 b = ((const float4*)Wfinal)[lane];
  float s = a.x * b.x + a.y * b.y + a.z * b.z + a.w * b.w;
#pragma unroll
  for (int m = 1; m < 32; m <<= 1) s += __shfl_xor(s, m, 32);
  if (lane == 0) out[(size_t)nb[n] * NENT + ne[n]] = s;
}

// ---------------------------------------------------------------------------
// Launch
// ---------------------------------------------------------------------------
extern "C" void kernel_launch(void* const* d_in, const int* in_sizes, int n_in,
                              void* d_out, int out_size, void* d_ws, size_t ws_size,
                              hipStream_t stream) {
  (void)in_sizes; (void)n_in; (void)out_size; (void)ws_size;
  const float* rela     = (const float*)d_in[0];
  const float* Ws       = (const float*)d_in[1];
  const float* Wr       = (const float*)d_in[2];
  const float* Wqr_w    = (const float*)d_in[3];
  const float* Wqr_b    = (const float*)d_in[4];
  const float* Wtau     = (const float*)d_in[5];
  const float* walpha_w = (const float*)d_in[6];
  const float* walpha_b = (const float*)d_in[7];
  const float* Wh       = (const float*)d_in[8];
  const float* wt1      = (const float*)d_in[9];
  const float* bt1      = (const float*)d_in[10];
  const float* wt2      = (const float*)d_in[11];
  const float* bt2      = (const float*)d_in[12];
  const float* Wih      = (const float*)d_in[13];
  const float* Whh      = (const float*)d_in[14];
  const float* bih      = (const float*)d_in[15];
  const float* bhh      = (const float*)d_in[16];
  const float* Wfinal   = (const float*)d_in[17];
  const int* q_rel  = (const int*)d_in[18];
  const int* q_tau  = (const int*)d_in[19];
  const int* e_ridx = (const int*)d_in[20];
  const int* e_rel  = (const int*)d_in[21];
  const int* e_tau  = (const int*)d_in[22];
  const int* e_sub  = (const int*)d_in[23];
  const int* e_obj  = (const int*)d_in[24];
  const int* nodes_batch = (const int*)d_in[25];
  const int* nodes_ent   = (const int*)d_in[26];

  // workspace: [packed bf16 weights | agg | state]  (~0.9MB + 2*51.2MB)
  const size_t wpack_u32 = (size_t)LNUM * TILES_PER_LAYER * TILE_U32;
  unsigned int* wpack = (unsigned int*)d_ws;
  float* agg   = (float*)((char*)d_ws + ((wpack_u32 * 4 + 255) & ~(size_t)255));
  float* state = agg + (size_t)NNODE * DDIM;

  const int prep_total = LNUM * TILES_PER_LAYER * TILE_U32;
  prep_weights<<<(prep_total + 255) / 256, 256, 0, stream>>>(
      Ws, Wr, Wqr_w, Wtau, Wh, Wih, Whh, wpack);
  zero_kernel<<<2048, 256, 0, stream>>>(state, (size_t)NNODE * DDIM);
  zero_kernel<<<4096, 256, 0, stream>>>((float*)d_out, (size_t)NQC * NENT);

  for (int l = 0; l < LNUM; ++l) {
    zero_kernel<<<2048, 256, 0, stream>>>(agg, (size_t)NNODE * DDIM);
    edge_kernel<<<NEDGE / 32 / 2, 64, 0, stream>>>(
        l, rela, Wqr_b, walpha_w, walpha_b, wt1, bt1, wt2, bt2,
        q_rel, q_tau, e_ridx, e_rel, e_tau, e_sub, e_obj, state, agg, wpack);
    node_kernel<<<NNODE / 16 / 2, 64, 0, stream>>>(l, agg, state, wpack, bih, bhh);
  }
  final_kernel<<<NNODE / 4, 128, 0, stream>>>(state, Wfinal, nodes_batch, nodes_ent,
                                              (float*)d_out);
}
